// FactorizedVectorQuantizer_81664508166541
// MI455X (gfx1250) — matrix operand order repaired and checked by reading, not compiled, not run on hardware
//
#include <hip/hip_runtime.h>
#include <hip/hip_bf16.h>

typedef __attribute__((ext_vector_type(16))) _Float16 v16h;
typedef __attribute__((ext_vector_type(8)))  _Float16 v8h;
typedef __attribute__((ext_vector_type(8)))  float    v8f;
typedef __attribute__((ext_vector_type(4)))  unsigned int v4u;
typedef __attribute__((ext_vector_type(4)))  int      v4i;
typedef __attribute__((ext_vector_type(8)))  int      v8i;

#define NTOK   16384   // 16 * 32 * 32 tokens
#define EDIM   256
#define LAT    32
#define NE     8192
#define HW     1024    // 32*32
#define BETA   0.25f

#define OUT_ZQ   0
#define OUT_LOSS 4194304
#define OUT_PERP 4194305
#define OUT_CU   4194306
#define OUT_IDX  4194307

#define CHUNK  512           // codes staged in LDS per pass
#define ROWP   40            // padded halves per code row (80B, 16B aligned)
#define NCHUNK (NE / CHUNK)  // 16

// ---------------- kernel 0: zero the accumulators we reduce into ----------------
__global__ void vq_init(unsigned* __restrict__ counts, float* __restrict__ lossAcc) {
  int t = blockIdx.x * blockDim.x + threadIdx.x;
  if (t < NE) counts[t] = 0u;
  if (t == 0) *lossAcc = 0.f;
}

// ---------------- kernel P: codebook -> f16 copy + ||e||^2 (f32) ----------------
__global__ void vq_prep(const float* __restrict__ emb,
                        _Float16* __restrict__ embh,
                        float* __restrict__ enorm) {
  int c = blockIdx.x * blockDim.x + threadIdx.x;
  if (c >= NE) return;
  const float* e = emb + c * LAT;
  float s = 0.f;
#pragma unroll
  for (int j = 0; j < LAT; ++j) {
    float v = e[j];
    s += v * v;
    embh[c * LAT + j] = (_Float16)v;
  }
  enorm[c] = s;
}

// ---------------- kernel 1: down projection z[b,c,h,w] -> z_lat[N,32] ----------------
// 2 threads per token (c-range halves) -> 1024 waves machine-wide, LDS combine.
__global__ void vq_down(const float* __restrict__ z,
                        const float* __restrict__ w_down,
                        const float* __restrict__ b_down,
                        float* __restrict__ zlat,
                        _Float16* __restrict__ zlath) {
  __shared__ float wdS[EDIM * LAT];     // 32 KB
  __shared__ float partS[128 * 33];     // padded rows: no bank conflicts on combine
  const int tid = threadIdx.x;
#pragma unroll
  for (int j = 0; j < LAT; ++j) wdS[tid * LAT + j] = w_down[tid * LAT + j];
  __syncthreads();

  const int tl   = tid & 127;
  const int half = tid >> 7;
  const int t    = blockIdx.x * 128 + tl;
  const int b    = t >> 10;
  const int hw   = t & (HW - 1);
  const float* zp = z + (size_t)b * EDIM * HW + hw;

  float acc[LAT];
#pragma unroll
  for (int j = 0; j < LAT; ++j) acc[j] = 0.f;

  const int c0 = half * 128;
#pragma unroll 4
  for (int c = 0; c < 128; ++c) {
    float zv = zp[(size_t)(c0 + c) * HW];         // coalesced across adjacent tokens
    const float* wr = &wdS[(c0 + c) * LAT];
#pragma unroll
    for (int j = 0; j < LAT; ++j) acc[j] = fmaf(zv, wr[j], acc[j]);
  }
  if (half == 1) {
#pragma unroll
    for (int j = 0; j < LAT; ++j) partS[tl * 33 + j] = acc[j];
  }
  __syncthreads();
  if (half == 0) {
#pragma unroll
    for (int j = 0; j < LAT; ++j) {
      float v = acc[j] + partS[tl * 33 + j] + b_down[j];
      zlat[t * LAT + j]  = v;
      zlath[t * LAT + j] = (_Float16)v;
    }
  }
}

// ---------------- TDM: one descriptor moves a 512x32-half codebook chunk to LDS ----------------
// Hardware padding: after every 16 DWORDs (one 32-half row) insert 4 DWORDs -> 40-half LDS rows.
__device__ __forceinline__ void tdm_load_codes(_Float16* dstLds, const _Float16* src) {
  unsigned lds = (unsigned)(uintptr_t)dstLds;          // low 32 bits of generic ptr = LDS offset
  unsigned long long ga = (unsigned long long)(uintptr_t)src;
  v4u g0;
  g0[0] = 1u;                                          // count=1, user descriptor
  g0[1] = lds;                                         // lds_addr
  g0[2] = (unsigned)(ga & 0xFFFFFFFFu);                // global_addr[31:0]
  g0[3] = (unsigned)((ga >> 32) & 0x1FFFFFFull) | (2u << 30);  // global_addr[56:32] | type=2
  v8i g1;
  g1[0] = (int)((1u << 16)     // data_size = 2 bytes
              | (1u << 20)     // pad_enable
              | (3u << 22)     // pad_interval: 16 DWORDs
              | (3u << 25));   // pad_amount : 4 DWORDs
  g1[1] = (int)(32u << 16);                 // tensor_dim0 = 32 (bits 63:48)
  g1[2] = (int)(((unsigned)NE) << 16);      // tensor_dim0_hi=0 | tensor_dim1 = 8192
  g1[3] = (int)(32u << 16);                 // tensor_dim1_hi=0 | tile_dim0 = 32
  g1[4] = (int)CHUNK;                       // tile_dim1 = 512 | tile_dim2 = 0
  g1[5] = (int)32;                          // tensor_dim0_stride = 32 elements
  g1[6] = 0;
  g1[7] = 0;
  v4i g2 = {0, 0, 0, 0};
  v4i g3 = {0, 0, 0, 0};
#if defined(__clang_major__) && (__clang_major__ >= 23)
  v8i g4 = {0, 0, 0, 0, 0, 0, 0, 0};
  __builtin_amdgcn_tensor_load_to_lds(g0, g1, g2, g3, g4, 0);
#else
  __builtin_amdgcn_tensor_load_to_lds(g0, g1, g2, g3, 0);
#endif
}

// ---------------- kernel 2: WMMA nearest-code search, TDM double-buffered ----------------
// block = 256 threads = 8 waves, 16 tokens/wave -> 128 tokens/block, grid = 128
__global__ void vq_search(const _Float16* __restrict__ zlath,
                          const _Float16* __restrict__ embh,
                          const float* __restrict__ enorm,
                          int* __restrict__ idxw) {
  __shared__ __align__(16) _Float16 embS[2][CHUNK * ROWP];  // 2 x 40 KB
  __shared__ float enS[2][CHUNK];                           // 4 KB

  const int tid    = threadIdx.x;
  const int lane   = tid & 31;
  const int wave   = tid >> 5;
  const int lx     = lane & 15;
  const int laneHi = lane >> 4;
  const int t0     = blockIdx.x * 128 + wave * 16;

  // --- A operand: 16 tokens x K=32 f16 (ISA 16-bit A 16x32 layout) ---
  const _Float16* pa = zlath + (t0 + lx) * LAT + laneHi * 8;
  v8h a0 = *(const v8h*)(pa);
  v8h a1 = *(const v8h*)(pa + 16);
  v16h aa;
#pragma unroll
  for (int i = 0; i < 8; ++i) { aa[i] = a0[i]; aa[8 + i] = a1[i]; }

  // prologue: TDM chunk 0 + stage its norms
  if (wave == 0) tdm_load_codes(embS[0], embh);
  enS[0][tid]       = enorm[tid];
  enS[0][tid + 256] = enorm[tid + 256];
  if (wave == 0) __builtin_amdgcn_s_wait_tensorcnt(0);
  __syncthreads();

  float bd[8];
  int   bi[8];
#pragma unroll
  for (int r = 0; r < 8; ++r) { bd[r] = 3.0e38f; bi[r] = 0; }

  for (int chunk = 0; chunk < NCHUNK; ++chunk) {
    const int cur = chunk & 1;
    if (chunk + 1 < NCHUNK) {                       // prefetch next chunk while computing
      if (wave == 0)
        tdm_load_codes(embS[cur ^ 1], embh + (size_t)(chunk + 1) * CHUNK * LAT);
      const int nb = (chunk + 1) * CHUNK;
      enS[cur ^ 1][tid]       = enorm[nb + tid];
      enS[cur ^ 1][tid + 256] = enorm[nb + tid + 256];
    }

    const _Float16* eb = embS[cur];
    const float*    en = enS[cur];
#pragma unroll 2
    for (int tile = 0; tile < CHUNK / 16; ++tile) {
      const int crow = tile * 16 + lx;
      // --- B operand: 16 codes x K=32; lanes 0-15 hold K0-7,16-23, lanes 16-31 K8-15,24-31 ---
      const _Float16* pb = eb + crow * ROWP + laneHi * 8;
      v8h b0 = *(const v8h*)(pb);
      v8h b1 = *(const v8h*)(pb + 16);
      v16h bb;
#pragma unroll
      for (int i = 0; i < 8; ++i) { bb[i] = b0[i]; bb[8 + i] = b1[i]; }

      v8f acc = {0.f, 0.f, 0.f, 0.f, 0.f, 0.f, 0.f, 0.f};
      acc = __builtin_amdgcn_wmma_f32_16x16x32_f16(
          false, aa, false, bb, (short)0, acc, false, false);

      const float e2 = en[crow];
      const int   ci = chunk * CHUNK + crow;        // ascending per lane -> first-min kept
#pragma unroll
      for (int r = 0; r < 8; ++r) {
        float d = fmaf(-2.f, acc[r], e2);           // ||e||^2 - 2 <z,e>
        if (d < bd[r]) { bd[r] = d; bi[r] = ci; }
      }
    }

    if (wave == 0 && chunk + 1 < NCHUNK) __builtin_amdgcn_s_wait_tensorcnt(0);
    __syncthreads();
  }

  // min+index reduce across the 16 lanes of each half (offsets stay within the half)
#pragma unroll
  for (int off = 8; off >= 1; off >>= 1) {
#pragma unroll
    for (int r = 0; r < 8; ++r) {
      float od = __shfl_xor(bd[r], off, 32);
      int   oi = __shfl_xor(bi[r], off, 32);
      if (od < bd[r] || (od == bd[r] && oi < bi[r])) { bd[r] = od; bi[r] = oi; }
    }
  }
  if (lx == 0) {
    int tb = t0 + laneHi * 8;                       // lanes<16 -> M=r ; lanes>=16 -> M=r+8
#pragma unroll
    for (int r = 0; r < 8; ++r) idxw[tb + r] = bi[r];
  }
}

// ---------------- kernel 3: up projection + loss + counts + idx out ----------------
// 2 threads per token (output-channel halves) -> 1024 waves machine-wide.
__global__ void vq_up(const int* __restrict__ idxw,
                      const float* __restrict__ emb,
                      const float* __restrict__ zlat,
                      const float* __restrict__ w_up,
                      const float* __restrict__ b_up,
                      float* __restrict__ out,
                      unsigned* __restrict__ counts,
                      float* __restrict__ lossAcc) {
  __shared__ float wuS[LAT * EDIM];   // 32 KB, [j*256 + c]
  const int tid = threadIdx.x;
#pragma unroll
  for (int j = 0; j < LAT; ++j) wuS[j * EDIM + tid] = w_up[j * EDIM + tid];
  __syncthreads();

  const int tl = tid & 127;
  const int ch = tid >> 7;
  const int t  = blockIdx.x * 128 + tl;
  const int b  = t >> 10;
  const int hw = t & (HW - 1);

  const int id = idxw[t];
  float q[LAT];
#pragma unroll
  for (int j = 0; j < LAT; ++j) q[j] = emb[id * LAT + j];

  if (ch == 0) {
    float lsum = 0.f;
#pragma unroll
    for (int j = 0; j < LAT; ++j) {
      float dzl = q[j] - zlat[t * LAT + j];
      lsum = fmaf(dzl, dzl, lsum);
    }
    atomicAdd(&counts[id], 1u);
    out[OUT_IDX + t] = (float)id;
#pragma unroll
    for (int off = 16; off >= 1; off >>= 1) lsum += __shfl_xor(lsum, off, 32);
    if ((tid & 31) == 0) atomicAdd(lossAcc, lsum);
  }

  float* op = out + OUT_ZQ + (size_t)b * EDIM * HW + hw;
  const int c0 = ch * 128;
#pragma unroll 4
  for (int c = 0; c < 128; ++c) {
    const int cc = c0 + c;
    float s = b_up[cc];
    const float* wc = &wuS[cc];
#pragma unroll
    for (int j = 0; j < LAT; ++j) s = fmaf(q[j], wc[j * EDIM], s);
    op[(size_t)cc * HW] = s;                        // coalesced across tokens
  }
}

// ---------------- kernel 4: perplexity / cluster_use / loss scalar ----------------
__global__ void vq_stats(const unsigned* __restrict__ counts,
                         const float* __restrict__ lossAcc,
                         float* __restrict__ out) {
  __shared__ float sH[256];
  __shared__ int   sC[256];
  int tid = threadIdx.x;
  float H = 0.f;
  int   cu = 0;
  for (int c = tid; c < NE; c += 256) {
    float p = (float)counts[c] * (1.0f / (float)NTOK);
    H += p * logf(p + 1e-10f);          // p==0 contributes exactly 0
    cu += (p > 0.f) ? 1 : 0;
  }
  sH[tid] = H; sC[tid] = cu;
  __syncthreads();
  for (int s = 128; s > 0; s >>= 1) {
    if (tid < s) { sH[tid] += sH[tid + s]; sC[tid] += sC[tid + s]; }
    __syncthreads();
  }
  if (tid == 0) {
    out[OUT_LOSS] = (1.0f + BETA) * (*lossAcc) / (float)(NTOK * LAT);
    out[OUT_PERP] = expf(-sH[0]);
    out[OUT_CU]   = (float)sC[0];
  }
}

// ---------------- launch ----------------
extern "C" void kernel_launch(void* const* d_in, const int* in_sizes, int n_in,
                              void* d_out, int out_size, void* d_ws, size_t ws_size,
                              hipStream_t stream) {
  const float* z      = (const float*)d_in[0];
  const float* w_down = (const float*)d_in[1];
  const float* b_down = (const float*)d_in[2];
  const float* w_up   = (const float*)d_in[3];
  const float* b_up   = (const float*)d_in[4];
  const float* emb    = (const float*)d_in[5];
  float* out = (float*)d_out;

  char* w = (char*)d_ws;
  float*    zlat    = (float*)w;      w += (size_t)NTOK * LAT * 4;
  _Float16* zlath   = (_Float16*)w;   w += (size_t)NTOK * LAT * 2;
  _Float16* embh    = (_Float16*)w;   w += (size_t)NE * LAT * 2;
  float*    enorm   = (float*)w;      w += (size_t)NE * 4;
  int*      idxw    = (int*)w;        w += (size_t)NTOK * 4;
  unsigned* counts  = (unsigned*)w;   w += (size_t)NE * 4;
  float*    lossAcc = (float*)w;      w += 256;

  vq_init  <<<NE / 256, 256, 0, stream>>>(counts, lossAcc);
  vq_prep  <<<NE / 256, 256, 0, stream>>>(emb, embh, enorm);
  vq_down  <<<NTOK / 128, 256, 0, stream>>>(z, w_down, b_down, zlat, zlath);
  vq_search<<<NTOK / 128, 256, 0, stream>>>(zlath, embh, enorm, idxw);
  vq_up    <<<NTOK / 128, 256, 0, stream>>>(idxw, emb, zlat, w_up, b_up, out, counts, lossAcc);
  vq_stats <<<1, 256, 0, stream>>>(counts, lossAcc, out);
}
